// SSAAttention_21741124453061
// MI455X (gfx1250) — compile-verified, matrix operand order checked by reading, not run on hardware
//
#include <hip/hip_runtime.h>
#include <hip/hip_bf16.h>
#include <stdint.h>

// ---------------- problem constants ----------------
constexpr int S_LEN = 4096;    // sequence
constexpr int HIDN  = 4096;    // hidden
constexpr int NHEAD = 32;
constexpr int NKVH  = 8;
constexpr int HDIM  = 128;
constexpr int NQKV  = 6144;    // q(4096) | k(1024) | v(1024) fused N
constexpr int NLMK  = 64;      // landmarks
constexpr int WIN   = 192;     // local window width (3 * 64)
constexpr float ATT_SCALE = 0.08838834764831845f; // 1/sqrt(128)

// ---------------- WMMA types ----------------
typedef __attribute__((ext_vector_type(16))) __bf16 v16bf;
typedef __attribute__((ext_vector_type(8)))  float  v8f;
typedef __attribute__((ext_vector_type(2)))  float  v2f;

union BF16x16 { v16bf v; unsigned short u[16]; uint4 q[2]; };

// ---------------- CDNA5 async global->LDS helpers ----------------
__device__ __forceinline__ void async_g2l_b128(void* lds_dst, const void* gsrc)
{
    // VDST = LDS byte address (wave-relative == low 32 bits of generic ptr),
    // VADDR = 64-bit global address, saddr = off.  Tracked by ASYNCcnt.
    unsigned int l = (unsigned int)(unsigned long long)lds_dst;
    asm volatile("global_load_async_to_lds_b128 %0, %1, off"
                 :: "v"(l), "v"(gsrc)
                 : "memory");
}
__device__ __forceinline__ void wait_async0()
{
    asm volatile("s_wait_asynccnt 0x0" ::: "memory");
}

// ---------------- f32 -> bf16 hi/lo split ----------------
__global__ void split_bf16_kernel(const float* __restrict__ src,
                                  unsigned short* __restrict__ hi,
                                  unsigned short* __restrict__ lo,
                                  int rows, int cols, int ld_dst, int col_off)
{
    int idx = blockIdx.x * 256 + threadIdx.x;
    if (idx >= rows * cols) return;
    int r = idx / cols, c = idx - r * cols;
    float x = src[(size_t)r * cols + c];
    unsigned int b  = __float_as_uint(x);
    unsigned int rh = (b + 0x7FFFu + ((b >> 16) & 1u)) >> 16;   // RNE to bf16
    float hf = __uint_as_float(rh << 16);
    float l  = x - hf;
    unsigned int b2 = __float_as_uint(l);
    unsigned int rl = (b2 + 0x7FFFu + ((b2 >> 16) & 1u)) >> 16;
    size_t o = (size_t)r * ld_dst + col_off + c;
    hi[o] = (unsigned short)rh;
    lo[o] = (unsigned short)rl;
}

// ---------------- bf16x3 GEMM: C(f32) = A(f32 as hi+lo) @ B(f32 as hi+lo) ----------------
constexpr int BM = 128, BN = 128, BK = 32;
constexpr int LDSA = BK + 8;   // ushort stride, A tile [BM][BK] row-major
constexpr int LDTB = BK + 8;   // ushort stride, B tile [BN][BK] TRANSPOSED (n-major)

__global__ __launch_bounds__(256) void gemm_bf16x3_kernel(
    const unsigned short* __restrict__ Ah, const unsigned short* __restrict__ Al,
    const unsigned short* __restrict__ Bh, const unsigned short* __restrict__ Bl,
    float* __restrict__ C, int M, int N, int K)
{
    __shared__ __align__(16) unsigned short sAh[BM * LDSA];
    __shared__ __align__(16) unsigned short sAl[BM * LDSA];
    __shared__ __align__(16) unsigned short sBh[BN * LDTB];   // [n][k]
    __shared__ __align__(16) unsigned short sBl[BN * LDTB];   // [n][k]

    const int m0 = blockIdx.y * BM, n0 = blockIdx.x * BN;
    const int t = threadIdx.x, lane = t & 31, wave = t >> 5;
    const int half = lane >> 4, lr = lane & 15;
    const int wm = wave >> 1, wn = wave & 1;   // 4x2 wave grid -> 32x64 per wave

    const int ar = t >> 1, ac = (t & 1) * 16;  // A tile: 128 rows x 32 k
    const int bkp = t >> 4;                    // 0..15: k-pair index (k = 2*bkp, 2*bkp+1)
    const int bng = (t & 15) * 8;              // n base (8 columns per thread)

    v8f z8 = {0,0,0,0,0,0,0,0};
    v8f acc[2][4];
    for (int i = 0; i < 2; i++)
        for (int j = 0; j < 4; j++) acc[i][j] = z8;

    for (int k0 = 0; k0 < K; k0 += BK) {
        // ---- A tiles: async DMA straight into LDS (ASYNCcnt) ----
        const unsigned short* gAh = Ah + (size_t)(m0 + ar) * K + k0 + ac;
        const unsigned short* gAl = Al + (size_t)(m0 + ar) * K + k0 + ac;
        async_g2l_b128(&sAh[ar * LDSA + ac],     gAh);
        async_g2l_b128(&sAh[ar * LDSA + ac + 8], gAh + 8);
        async_g2l_b128(&sAl[ar * LDSA + ac],     gAl);
        async_g2l_b128(&sAl[ar * LDSA + ac + 8], gAl + 8);

        // ---- B tiles: register-path transpose into [n][k] LDS layout ----
        const unsigned short* gBh0 = Bh + (size_t)(k0 + 2 * bkp) * N + n0 + bng;
        const unsigned short* gBl0 = Bl + (size_t)(k0 + 2 * bkp) * N + n0 + bng;
        union { uint4 q; unsigned short u[8]; } r0h, r1h, r0l, r1l;
        r0h.q = *(const uint4*)gBh0;
        r1h.q = *(const uint4*)(gBh0 + N);
        r0l.q = *(const uint4*)gBl0;
        r1l.q = *(const uint4*)(gBl0 + N);
        #pragma unroll
        for (int i = 0; i < 8; i++) {
            unsigned int ph = (unsigned int)r0h.u[i] | ((unsigned int)r1h.u[i] << 16);
            unsigned int pl = (unsigned int)r0l.u[i] | ((unsigned int)r1l.u[i] << 16);
            *(unsigned int*)&sBh[(bng + i) * LDTB + 2 * bkp] = ph;
            *(unsigned int*)&sBl[(bng + i) * LDTB + 2 * bkp] = pl;
        }

        if (k0 + BK < K) {   // global_prefetch_b8 for next k-tile
            __builtin_prefetch(gAh + BK, 0, 1);
            __builtin_prefetch(gBh0 + (size_t)BK * N, 0, 1);
        }

        wait_async0();          // this wave's async A-tile copies done
        __syncthreads();        // block-wide: LDS tiles ready

        // ---- fragments: every operand is two ds_load_b128 ----
        BF16x16 a_hi[2], a_lo[2];
        #pragma unroll
        for (int mt = 0; mt < 2; mt++) {
            const int row = wm * 32 + mt * 16 + lr;
            const unsigned short* pah = &sAh[row * LDSA + 8 * half];
            const unsigned short* pal = &sAl[row * LDSA + 8 * half];
            a_hi[mt].q[0] = *(const uint4*)pah;
            a_hi[mt].q[1] = *(const uint4*)(pah + 16);
            a_lo[mt].q[0] = *(const uint4*)pal;
            a_lo[mt].q[1] = *(const uint4*)(pal + 16);
        }
        #pragma unroll
        for (int nt = 0; nt < 4; nt++) {
            const int col = wn * 64 + nt * 16 + lr;
            const unsigned short* pbh = &sBh[col * LDTB + 8 * half];
            const unsigned short* pbl = &sBl[col * LDTB + 8 * half];
            BF16x16 b_hi, b_lo;
            b_hi.q[0] = *(const uint4*)pbh;
            b_hi.q[1] = *(const uint4*)(pbh + 16);
            b_lo.q[0] = *(const uint4*)pbl;
            b_lo.q[1] = *(const uint4*)(pbl + 16);
            #pragma unroll
            for (int mt = 0; mt < 2; mt++) {
                acc[mt][nt] = __builtin_amdgcn_wmma_f32_16x16x32_bf16(
                    false, a_hi[mt].v, false, b_hi.v, (short)0, acc[mt][nt], false, false);
                acc[mt][nt] = __builtin_amdgcn_wmma_f32_16x16x32_bf16(
                    false, a_hi[mt].v, false, b_lo.v, (short)0, acc[mt][nt], false, false);
                acc[mt][nt] = __builtin_amdgcn_wmma_f32_16x16x32_bf16(
                    false, a_lo[mt].v, false, b_hi.v, (short)0, acc[mt][nt], false, false);
            }
        }
        __syncthreads();        // before next iteration overwrites LDS
    }

    #pragma unroll
    for (int mt = 0; mt < 2; mt++)
        #pragma unroll
        for (int nt = 0; nt < 4; nt++)
            #pragma unroll
            for (int i = 0; i < 8; i++) {
                const int gr = m0 + wm * 32 + mt * 16 + i + 8 * half;
                const int gc = n0 + wn * 64 + nt * 16 + lr;
                C[(size_t)gr * N + gc] = acc[mt][nt][i];
            }
}

// ---------------- RoPE (in place on qkv columns) ----------------
__global__ void rope_kernel(float* __restrict__ x, int nheads, int ld)
{
    int idx = blockIdx.x * 256 + threadIdx.x;
    int total = S_LEN * nheads * 64;
    if (idx >= total) return;
    int j = idx & 63;
    int tmp = idx >> 6;
    int hh = tmp % nheads;
    int s  = tmp / nheads;
    // inv_freq = 10000^(-j/64) = exp(-j * ln(10000)/64)
    float ang = (float)s * __expf(-0.14391156831f * (float)j);
    float sn, cs;
    __sincosf(ang, &sn, &cs);
    float* p = x + (size_t)s * ld + hh * 128 + j;
    float x1 = p[0], x2 = p[64];
    p[0]  = x1 * cs - x2 * sn;
    p[64] = x2 * cs + x1 * sn;
}

// ---------------- SSA attention: local band + landmarks ----------------
// qkv layout: [S][6144] = q(32 heads x128) | k(8 kv x128) | v(8 kv x128)
// grid: (S/64, NHEAD); 8 waves: 0-3 local stripes, 4-7 landmark stripes.
__global__ __launch_bounds__(256) void ssa_attn_kernel(const float* __restrict__ qkv,
                                                       float* __restrict__ ctx)
{
    __shared__ __align__(16) float s_w [4][16 * WIN];   // local softmax weights
    __shared__ __align__(16) float s_lw[4][16 * NLMK];  // landmark softmax weights
    __shared__ __align__(16) float s_ctx[64 * HDIM];    // 0.3 * global_out staging

    const int blk = blockIdx.x;
    const int h   = blockIdx.y;
    const int kvh = h >> 2;                 // GQA: 4 q-heads per kv-head
    const int t = threadIdx.x, wave = t >> 5, lane = t & 31;
    const int half = lane >> 4, lr = lane & 15;
    const int s0 = blk * 64;
    const int sr = (wave & 3) * 16;         // 16-row stripe within the 64-row block

    const float* qp = qkv + (size_t)h * HDIM;
    const float* kp = qkv + 4096 + (size_t)kvh * HDIM;
    const float* vp = qkv + 5120 + (size_t)kvh * HDIM;

    const v8f z8 = {0,0,0,0,0,0,0,0};

    // ---- phase 1: scores + softmax -> LDS ----
    if (wave < 4) {
        v8f sc[12];
        #pragma unroll
        for (int i = 0; i < 12; i++) sc[i] = z8;
        for (int kk = 0; kk < 32; kk++) {           // K = 128 in steps of 4
            const int kd0 = kk * 4 + 2 * half;
            const size_t qrow = (size_t)(s0 + sr + lr) * NQKV;
            v2f a; a[0] = qp[qrow + kd0]; a[1] = qp[qrow + kd0 + 1];
            #pragma unroll
            for (int nt = 0; nt < 12; nt++) {
                const int j = nt * 16 + lr;         // window key index 0..191
                const int kpos = s0 - 64 + j;
                v2f b = {0.0f, 0.0f};
                if (kpos >= 0 && kpos < S_LEN) {
                    const size_t kr = (size_t)kpos * NQKV;
                    b[0] = kp[kr + kd0]; b[1] = kp[kr + kd0 + 1];
                }
                sc[nt] = __builtin_amdgcn_wmma_f32_16x16x4_f32(
                    false, a, false, b, (short)0, sc[nt], false, false);
            }
        }
        float mx[8], sm[8];
        #pragma unroll
        for (int i = 0; i < 8; i++) mx[i] = -3.0e38f;
        #pragma unroll
        for (int nt = 0; nt < 12; nt++) {
            const int j = nt * 16 + lr;
            #pragma unroll
            for (int i = 0; i < 8; i++) {
                const int ib = sr + i + 8 * half;   // query row within block
                float v = sc[nt][i] * ATT_SCALE;
                v = (j >= ib && j <= ib + 64) ? v : -3.0e38f;
                sc[nt][i] = v;
                mx[i] = fmaxf(mx[i], v);
            }
        }
        #pragma unroll
        for (int i = 0; i < 8; i++) {
            for (int d = 1; d < 16; d <<= 1) mx[i] = fmaxf(mx[i], __shfl_xor(mx[i], d, 32));
            sm[i] = 0.0f;
        }
        #pragma unroll
        for (int nt = 0; nt < 12; nt++)
            #pragma unroll
            for (int i = 0; i < 8; i++) {
                float e = __expf(sc[nt][i] - mx[i]);
                sc[nt][i] = e; sm[i] += e;
            }
        #pragma unroll
        for (int i = 0; i < 8; i++) {
            for (int d = 1; d < 16; d <<= 1) sm[i] += __shfl_xor(sm[i], d, 32);
            sm[i] = 1.0f / sm[i];
        }
        #pragma unroll
        for (int nt = 0; nt < 12; nt++)
            #pragma unroll
            for (int i = 0; i < 8; i++)
                s_w[wave][(i + 8 * half) * WIN + nt * 16 + lr] = sc[nt][i] * sm[i];
    } else {
        v8f sc[4];
        #pragma unroll
        for (int i = 0; i < 4; i++) sc[i] = z8;
        for (int kk = 0; kk < 32; kk++) {
            const int kd0 = kk * 4 + 2 * half;
            const size_t qrow = (size_t)(s0 + sr + lr) * NQKV;
            v2f a; a[0] = qp[qrow + kd0]; a[1] = qp[qrow + kd0 + 1];
            #pragma unroll
            for (int nt = 0; nt < 4; nt++) {
                const int j = nt * 16 + lr;         // landmark index 0..63
                const size_t kr = (size_t)(j * 64) * NQKV;  // landmark key pos = 64*j
                v2f b; b[0] = kp[kr + kd0]; b[1] = kp[kr + kd0 + 1];
                sc[nt] = __builtin_amdgcn_wmma_f32_16x16x4_f32(
                    false, a, false, b, (short)0, sc[nt], false, false);
            }
        }
        float mx[8], sm[8];
        #pragma unroll
        for (int i = 0; i < 8; i++) mx[i] = -3.0e38f;
        #pragma unroll
        for (int nt = 0; nt < 4; nt++) {
            const int j = nt * 16 + lr;
            #pragma unroll
            for (int i = 0; i < 8; i++) {
                float v = sc[nt][i] * ATT_SCALE;
                v = (j <= blk) ? v : -3.0e38f;      // causal over landmarks
                sc[nt][i] = v;
                mx[i] = fmaxf(mx[i], v);
            }
        }
        #pragma unroll
        for (int i = 0; i < 8; i++) {
            for (int d = 1; d < 16; d <<= 1) mx[i] = fmaxf(mx[i], __shfl_xor(mx[i], d, 32));
            sm[i] = 0.0f;
        }
        #pragma unroll
        for (int nt = 0; nt < 4; nt++)
            #pragma unroll
            for (int i = 0; i < 8; i++) {
                float e = __expf(sc[nt][i] - mx[i]);
                sc[nt][i] = e; sm[i] += e;
            }
        #pragma unroll
        for (int i = 0; i < 8; i++) {
            for (int d = 1; d < 16; d <<= 1) sm[i] += __shfl_xor(sm[i], d, 32);
            sm[i] = 1.0f / sm[i];
        }
        #pragma unroll
        for (int nt = 0; nt < 4; nt++)
            #pragma unroll
            for (int i = 0; i < 8; i++)
                s_lw[wave - 4][(i + 8 * half) * NLMK + nt * 16 + lr] = sc[nt][i] * sm[i];
    }
    __syncthreads();

    // ---- phase 2: P @ V ----
    v8f out[8];
    #pragma unroll
    for (int i = 0; i < 8; i++) out[i] = z8;

    if (wave < 4) {
        for (int kk = 0; kk < 48; kk++) {           // K = 192 window keys
            const int ki0 = kk * 4 + 2 * half;
            v2f a; a[0] = s_w[wave][lr * WIN + ki0]; a[1] = s_w[wave][lr * WIN + ki0 + 1];
            const int kp0 = s0 - 64 + ki0;
            #pragma unroll
            for (int nt = 0; nt < 8; nt++) {
                const int d = nt * 16 + lr;
                v2f b = {0.0f, 0.0f};
                if (kp0 >= 0 && kp0 < S_LEN)         b[0] = vp[(size_t)kp0 * NQKV + d];
                if (kp0 + 1 >= 0 && kp0 + 1 < S_LEN) b[1] = vp[(size_t)(kp0 + 1) * NQKV + d];
                out[nt] = __builtin_amdgcn_wmma_f32_16x16x4_f32(
                    false, a, false, b, (short)0, out[nt], false, false);
            }
        }
    } else {
        for (int kk = 0; kk < 16; kk++) {           // K = 64 landmarks
            const int ki0 = kk * 4 + 2 * half;
            v2f a; a[0] = s_lw[wave - 4][lr * NLMK + ki0]; a[1] = s_lw[wave - 4][lr * NLMK + ki0 + 1];
            #pragma unroll
            for (int nt = 0; nt < 8; nt++) {
                const int d = nt * 16 + lr;
                v2f b;
                b[0] = vp[(size_t)(ki0 * 64) * NQKV + d];
                b[1] = vp[(size_t)((ki0 + 1) * 64) * NQKV + d];
                out[nt] = __builtin_amdgcn_wmma_f32_16x16x4_f32(
                    false, a, false, b, (short)0, out[nt], false, false);
            }
        }
        #pragma unroll
        for (int nt = 0; nt < 8; nt++)
            #pragma unroll
            for (int i = 0; i < 8; i++)
                s_ctx[(sr + i + 8 * half) * HDIM + nt * 16 + lr] = 0.3f * out[nt][i];
    }
    __syncthreads();

    if (wave < 4) {
        #pragma unroll
        for (int nt = 0; nt < 8; nt++)
            #pragma unroll
            for (int i = 0; i < 8; i++) {
                const int r = sr + i + 8 * half;
                const int d = nt * 16 + lr;
                float o = 0.7f * out[nt][i] + s_ctx[r * HDIM + d];
                ctx[(size_t)(s0 + r) * HIDN + (size_t)h * HDIM + d] = o;
            }
    }
}

// ---------------- host-side launcher ----------------
extern "C" void kernel_launch(void* const* d_in, const int* in_sizes, int n_in,
                              void* d_out, int out_size, void* d_ws, size_t ws_size,
                              hipStream_t stream)
{
    (void)in_sizes; (void)n_in; (void)out_size; (void)ws_size;
    const float* hs = (const float*)d_in[0];
    const float* wq = (const float*)d_in[1];
    const float* wk = (const float*)d_in[2];
    const float* wv = (const float*)d_in[3];
    const float* wo = (const float*)d_in[4];
    float* out = (float*)d_out;

    char* ws = (char*)d_ws;
    size_t off = 0;
    auto alloc = [&](size_t bytes) -> char* {
        char* p = ws + off;
        off += (bytes + 255) & ~(size_t)255;
        return p;
    };

    unsigned short* hs_hi   = (unsigned short*)alloc((size_t)S_LEN * HIDN * 2);
    unsigned short* hs_lo   = (unsigned short*)alloc((size_t)S_LEN * HIDN * 2);
    unsigned short* wqkv_hi = (unsigned short*)alloc((size_t)HIDN * NQKV * 2);
    unsigned short* wqkv_lo = (unsigned short*)alloc((size_t)HIDN * NQKV * 2);
    unsigned short* wo_hi   = (unsigned short*)alloc((size_t)HIDN * HIDN * 2);
    unsigned short* wo_lo   = (unsigned short*)alloc((size_t)HIDN * HIDN * 2);
    float* qkv = (float*)alloc((size_t)S_LEN * NQKV * 4);
    float* ctx = (float*)alloc((size_t)S_LEN * HIDN * 4);
    // hs splits are dead after the QKV GEMM -> reuse for ctx splits
    unsigned short* ctx_hi = hs_hi;
    unsigned short* ctx_lo = hs_lo;

    // 1) split inputs to bf16 hi/lo
    {
        int tot = S_LEN * HIDN;
        split_bf16_kernel<<<(tot + 255) / 256, 256, 0, stream>>>(hs, hs_hi, hs_lo, S_LEN, HIDN, HIDN, 0);
        tot = HIDN * HIDN;
        split_bf16_kernel<<<(tot + 255) / 256, 256, 0, stream>>>(wq, wqkv_hi, wqkv_lo, HIDN, 4096, NQKV, 0);
        tot = HIDN * 1024;
        split_bf16_kernel<<<(tot + 255) / 256, 256, 0, stream>>>(wk, wqkv_hi, wqkv_lo, HIDN, 1024, NQKV, 4096);
        split_bf16_kernel<<<(tot + 255) / 256, 256, 0, stream>>>(wv, wqkv_hi, wqkv_lo, HIDN, 1024, NQKV, 5120);
        tot = HIDN * HIDN;
        split_bf16_kernel<<<(tot + 255) / 256, 256, 0, stream>>>(wo, wo_hi, wo_lo, HIDN, HIDN, HIDN, 0);
    }

    // 2) fused QKV projection: [4096 x 4096] @ [4096 x 6144]
    {
        dim3 grid(NQKV / BN, S_LEN / BM);
        gemm_bf16x3_kernel<<<grid, 256, 0, stream>>>(hs_hi, hs_lo, wqkv_hi, wqkv_lo,
                                                     qkv, S_LEN, NQKV, HIDN);
    }

    // 3) RoPE on q and k (in place)
    rope_kernel<<<(S_LEN * NHEAD * 64 + 255) / 256, 256, 0, stream>>>(qkv, NHEAD, NQKV);
    rope_kernel<<<(S_LEN * NKVH * 64 + 255) / 256, 256, 0, stream>>>(qkv + 4096, NKVH, NQKV);

    // 4) SSA attention -> ctx [S][4096]
    {
        dim3 grid(S_LEN / 64, NHEAD);
        ssa_attn_kernel<<<grid, 256, 0, stream>>>(qkv, ctx);
    }

    // 5) output projection: ctx @ wo -> out
    {
        int tot = S_LEN * HIDN;
        split_bf16_kernel<<<(tot + 255) / 256, 256, 0, stream>>>(ctx, ctx_hi, ctx_lo, S_LEN, HIDN, HIDN, 0);
        dim3 grid(HIDN / BN, S_LEN / BM);
        gemm_bf16x3_kernel<<<grid, 256, 0, stream>>>(ctx_hi, ctx_lo, wo_hi, wo_lo,
                                                     out, S_LEN, HIDN, HIDN);
    }
}
